// SwinTransformerBlock_78065325572354
// MI455X (gfx1250) — matrix operand order repaired and checked by reading, not compiled, not run on hardware
//
#include <hip/hip_runtime.h>

// Swin block for MI455X (gfx1250, wave32). All GEMMs via v_wmma_f32_16x16x32_f16.
// Tile staging uses GLOBAL_LOAD_ASYNC_TO_LDS_B128 + s_wait_asynccnt (double
// buffered) when the toolchain exposes the builtins; sync fallback otherwise.

typedef __attribute__((ext_vector_type(16))) _Float16 v16h;
typedef __attribute__((ext_vector_type(8)))  _Float16 h8;
typedef __attribute__((ext_vector_type(8)))  float    v8f;
typedef __attribute__((ext_vector_type(4)))  int      v4i;

#ifdef __has_builtin
#if __has_builtin(__builtin_amdgcn_global_load_async_to_lds_b128)
#define ATHENA_ASYNC_LDS 1
#endif
#if __has_builtin(__builtin_amdgcn_s_wait_asynccnt)
#define ATHENA_WAIT_BUILTIN 1
#endif
#endif
#ifndef ATHENA_ASYNC_LDS
#define ATHENA_ASYNC_LDS 0
#endif
#ifndef ATHENA_WAIT_BUILTIN
#define ATHENA_WAIT_BUILTIN 0
#endif

#if ATHENA_ASYNC_LDS
typedef __attribute__((address_space(1))) v4i gv4i_t;   // global int4
typedef __attribute__((address_space(3))) v4i lv4i_t;   // LDS int4
#endif

namespace {
constexpr int kDim   = 384;
constexpr int kHeads = 12;
constexpr int kHd    = 32;
constexpr int kWs    = 7;
constexpr int kNTok  = 49;                 // tokens per window
constexpr int kNw    = 64;                 // windows per image (8x8)
constexpr int kImgH  = 56;
constexpr int kImgW  = 56;
constexpr int kL     = kImgH * kImgW;
constexpr int kRows  = 32 * kNw * kNTok;   // 100352 (== 784 * 128)
constexpr int kQkvN  = 3 * kDim;           // 1152
constexpr int kMlpH  = 4 * kDim;           // 1536
constexpr int kShift = 3;
}

// 16-byte global -> LDS copy (async when available)
__device__ __forceinline__ void lds_cp16(const _Float16* g, _Float16* l) {
#if ATHENA_ASYNC_LDS
    __builtin_amdgcn_global_load_async_to_lds_b128(
        (gv4i_t*)(void*)g, (lv4i_t*)(void*)l, 0, 0);
#else
    *(h8*)l = *(const h8*)g;
#endif
}

template<int N>
__device__ __forceinline__ void wait_asynccnt() {
#if ATHENA_ASYNC_LDS
#if ATHENA_WAIT_BUILTIN
    __builtin_amdgcn_s_wait_asynccnt(N);
#else
    if constexpr (N == 0) asm volatile("s_wait_asynccnt 0x0" ::: "memory");
    else                  asm volatile("s_wait_asynccnt 0x4" ::: "memory");
#endif
#endif
}

__device__ __forceinline__ int swin_band(int p) {
    return p < (kImgH - kWs) ? 0 : (p < (kImgH - kShift) ? 1 : 2);
}

// ---------------------------------------------------------------------------
// Elementwise f32 -> f16 cast (weights keep their natural (N x K) layout,
// which is exactly the [n][k] layout the WMMA B-fragment loader wants).
// ---------------------------------------------------------------------------
__global__ void cast_f16_kernel(const float* __restrict__ w,
                                _Float16* __restrict__ o, int total) {
    int idx = blockIdx.x * 256 + threadIdx.x;
    if (idx < total) o[idx] = (_Float16)w[idx];
}

// ---------------------------------------------------------------------------
// LayerNorm -> f16. MODE 0: natural source row, windowed(+shift) dest row (LN1).
//                   MODE 1: identity rows (LN2).  One wave per row of 384.
// ---------------------------------------------------------------------------
template<int MODE>
__launch_bounds__(256)
__global__ void ln_to_f16_kernel(const float* __restrict__ x,
                                 const float* __restrict__ g,
                                 const float* __restrict__ b,
                                 _Float16* __restrict__ out) {
    int row  = (blockIdx.x * blockDim.x + threadIdx.x) >> 5;
    int lane = threadIdx.x & 31;
    if (row >= kRows) return;
    int src;
    if (MODE == 0) {
        int bi  = row / (kNw * kNTok);
        int rem = row - bi * (kNw * kNTok);
        int win = rem / kNTok;
        int n   = rem - win * kNTok;
        int wh = win >> 3, ww = win & 7;
        int i = n / kWs, j = n - i * kWs;
        int hh = (wh * kWs + i + kShift) % kImgH;
        int wc = (ww * kWs + j + kShift) % kImgW;
        src = bi * kL + hh * kImgW + wc;
    } else {
        src = row;
    }
    const float* xr = x + (size_t)src * kDim;
    float v[12];
    float s = 0.f;
#pragma unroll
    for (int i = 0; i < 12; ++i) { v[i] = xr[i * 32 + lane]; s += v[i]; }
#pragma unroll
    for (int m = 16; m; m >>= 1) s += __shfl_xor(s, m, 32);
    float mean = s * (1.0f / kDim);
    float vs = 0.f;
#pragma unroll
    for (int i = 0; i < 12; ++i) { float d = v[i] - mean; vs += d * d; }
#pragma unroll
    for (int m = 16; m; m >>= 1) vs += __shfl_xor(vs, m, 32);
    float rstd = rsqrtf(vs * (1.0f / kDim) + 1e-5f);
    _Float16* orow = out + (size_t)row * kDim;
#pragma unroll
    for (int i = 0; i < 12; ++i) {
        int ch = i * 32 + lane;
        orow[ch] = (_Float16)((v[i] - mean) * rstd * g[ch] + b[ch]);
    }
}

// ---------------------------------------------------------------------------
// WMMA GEMM: C(M x ncols) = A(M x KDIM f16 rm) @ Bw(ncols x KDIM f16 rm)^T + bias
// Block tile 128x128, 8 waves (4x2), each wave 32x64 = 2x4 C tiles, K step 32.
// Double-buffered async-LDS staging; both tiles are 128 rows x 32 K-contiguous.
// EPI: 0 = f16 out; 1 = exact GELU -> f16; 2 = proj: window-reverse + x residual
//      -> f32 natural layout; 3 = +resid f32 (same row) -> f32 out.
// ---------------------------------------------------------------------------
template<int KDIM, int EPI>
__launch_bounds__(256, 2)
__global__ void gemm_wmma_kernel(const _Float16* __restrict__ A,
                                 const _Float16* __restrict__ Bw,
                                 const float* __restrict__ bias,
                                 const float* __restrict__ resid,
                                 _Float16* __restrict__ outh,
                                 float* __restrict__ outf,
                                 int ncols) {
    __shared__ _Float16 As[2][128][40];  // [buf][m][k]
    __shared__ _Float16 Bs[2][128][40];  // [buf][n][k]

    const int tid  = threadIdx.x;
    const int wid  = tid >> 5;
    const int lane = tid & 31;
    const int half = lane >> 4;
    const int ln   = lane & 15;
    const int wm   = wid & 3;            // wave row (0..3)  -> 32 rows each
    const int wn   = wid >> 2;           // wave col (0..1)  -> 64 cols each
    const int blockM = blockIdx.x * 128;
    const int blockN = blockIdx.y * 128;

    // per-thread staging coordinates: 2 chunks per tile per matrix
    const int sr0 = tid >> 2, sc0 = (tid & 3) * 8;
    const int sr1 = (tid + 256) >> 2, sc1 = ((tid + 256) & 3) * 8;

    auto stage = [&](int buf, int k0) {
        lds_cp16(A  + (size_t)(blockM + sr0) * KDIM + (k0 + sc0), &As[buf][sr0][sc0]);
        lds_cp16(A  + (size_t)(blockM + sr1) * KDIM + (k0 + sc1), &As[buf][sr1][sc1]);
        lds_cp16(Bw + (size_t)(blockN + sr0) * KDIM + (k0 + sc0), &Bs[buf][sr0][sc0]);
        lds_cp16(Bw + (size_t)(blockN + sr1) * KDIM + (k0 + sc1), &Bs[buf][sr1][sc1]);
    };

    v8f acc[2][4];
#pragma unroll
    for (int mi = 0; mi < 2; ++mi)
#pragma unroll
        for (int ni = 0; ni < 4; ++ni)
#pragma unroll
            for (int e = 0; e < 8; ++e) acc[mi][ni][e] = 0.0f;

    stage(0, 0);
    int buf = 0;
#pragma unroll 1
    for (int k0 = 0; k0 < KDIM; k0 += 32) {
        const bool more = (k0 + 32 < KDIM);
        if (more) {
            stage(buf ^ 1, k0 + 32);     // prefetch next tile (stays in flight)
            wait_asynccnt<4>();          // current tile's 4 ops retired
        } else {
            wait_asynccnt<0>();
        }
        __syncthreads();

        // A fragments (16x32 f16): lanes 0-15 K0-7/16-23, lanes 16-31 K8-15/24-31
        v16h af[2];
#pragma unroll
        for (int mi = 0; mi < 2; ++mi) {
            int row = wm * 32 + mi * 16 + ln;
            h8 lo = *(const h8*)&As[buf][row][half * 8];
            h8 hi = *(const h8*)&As[buf][row][16 + half * 8];
#pragma unroll
            for (int e = 0; e < 8; ++e) { af[mi][e] = lo[e]; af[mi][8 + e] = hi[e]; }
        }
        // B fragments (32x16 f16): lanes 0-15 K0-7/16-23, lanes 16-31 K8-15/24-31
        v16h bf[4];
#pragma unroll
        for (int ni = 0; ni < 4; ++ni) {
            int col = wn * 64 + ni * 16 + ln;
            h8 lo = *(const h8*)&Bs[buf][col][half * 8];
            h8 hi = *(const h8*)&Bs[buf][col][16 + half * 8];
#pragma unroll
            for (int e = 0; e < 8; ++e) { bf[ni][e] = lo[e]; bf[ni][8 + e] = hi[e]; }
        }
#pragma unroll
        for (int mi = 0; mi < 2; ++mi)
#pragma unroll
            for (int ni = 0; ni < 4; ++ni)
                acc[mi][ni] = __builtin_amdgcn_wmma_f32_16x16x32_f16(
                    false, af[mi], false, bf[ni], (short)0, acc[mi][ni], false, false);
        __syncthreads();
        buf ^= 1;
    }

    // epilogue: C element e -> (M = e + 8*half, N = ln)
#pragma unroll
    for (int mi = 0; mi < 2; ++mi) {
#pragma unroll
        for (int ni = 0; ni < 4; ++ni) {
#pragma unroll
            for (int e = 0; e < 8; ++e) {
                int gr = blockM + wm * 32 + mi * 16 + e + half * 8;
                int gc = blockN + wn * 64 + ni * 16 + ln;
                float val = acc[mi][ni][e] + bias[gc];
                if constexpr (EPI == 0) {
                    outh[(size_t)gr * ncols + gc] = (_Float16)val;
                } else if constexpr (EPI == 1) {
                    val = 0.5f * val * (1.0f + erff(val * 0.70710678118654752f));
                    outh[(size_t)gr * ncols + gc] = (_Float16)val;
                } else if constexpr (EPI == 2) {
                    int bi  = gr / (kNw * kNTok);
                    int rem = gr - bi * (kNw * kNTok);
                    int win = rem / kNTok;
                    int n   = rem - win * kNTok;
                    int wh = win >> 3, ww = win & 7;
                    int i = n / kWs, j = n - i * kWs;
                    int hh = (wh * kWs + i + kShift) % kImgH;
                    int wc = (ww * kWs + j + kShift) % kImgW;
                    size_t nat = ((size_t)bi * kL + hh * kImgW + wc) * kDim + gc;
                    outf[nat] = val + resid[nat];
                } else {
                    size_t idx = (size_t)gr * ncols + gc;
                    outf[idx] = val + resid[idx];
                }
            }
        }
    }
}

// ---------------------------------------------------------------------------
// Windowed attention: one block per (window, head); 4 waves, each owns a 16-row
// strip of the 64x64 (padded 49x49) score matrix. K=32 -> single WMMA per tile.
// ---------------------------------------------------------------------------
__launch_bounds__(128, 2)
__global__ void attn_kernel(const _Float16* __restrict__ qkv,
                            const float* __restrict__ rpb,
                            _Float16* __restrict__ out) {
    __shared__ _Float16 Qs[64][40];      // [token][d]
    __shared__ _Float16 Ks[64][40];      // [token][d]
    __shared__ _Float16 Vt[32][72];      // [d][token] (K-contiguous for B frags)
    __shared__ _Float16 Am[64][72];      // softmaxed attention [row][col]
    __shared__ float    rpbh[169];       // rel-pos bias column for this head

    const int blk = blockIdx.x;
    const int h   = blk % kHeads;
    const int win = blk / kHeads;
    const int wi  = win % kNw;
    const int wh  = wi >> 3, ww = wi & 7;
    const size_t rowbase = (size_t)win * kNTok;
    const int tid = threadIdx.x;

    for (int i = tid; i < 169; i += 128) rpbh[i] = rpb[i * kHeads + h];

    // load Q/K/V (49x32 each, pad to 64 rows with zeros). Q/K async, V sync
    // (V is transposed into LDS so WMMA B fragments read K-contiguous).
    for (int v = tid; v < 256; v += 128) {
        int r = v >> 2;
        int c = (v & 3) * 8;
        if (r < kNTok) {
            const _Float16* rp = qkv + (rowbase + r) * kQkvN + h * kHd + c;
            lds_cp16(rp, &Qs[r][c]);
            lds_cp16(rp + kDim, &Ks[r][c]);
            h8 vv = *(const h8*)(rp + 2 * kDim);
#pragma unroll
            for (int e = 0; e < 8; ++e) Vt[c + e][r] = vv[e];
        } else {
            h8 z;
#pragma unroll
            for (int e = 0; e < 8; ++e) z[e] = (_Float16)0;
            *(h8*)&Qs[r][c] = z;
            *(h8*)&Ks[r][c] = z;
#pragma unroll
            for (int e = 0; e < 8; ++e) Vt[c + e][r] = (_Float16)0;
        }
    }
    wait_asynccnt<0>();
    __syncthreads();

    const int wv   = tid >> 5;
    const int lane = tid & 31;
    const int half = lane >> 4;
    const int ln   = lane & 15;
    const int strip = wv * 16;

    // --- stage 1: S = Q @ K^T, one wmma per 16x16 tile (K = 32) ---
    v16h aq;
    {
        int row = strip + ln;
        h8 lo = *(const h8*)&Qs[row][half * 8];
        h8 hi = *(const h8*)&Qs[row][16 + half * 8];
#pragma unroll
        for (int e = 0; e < 8; ++e) { aq[e] = lo[e]; aq[8 + e] = hi[e]; }
    }
    v8f sc[4];
#pragma unroll
    for (int t = 0; t < 4; ++t) {
        int col = t * 16 + ln;                   // key token
        v16h bk;
        h8 lo = *(const h8*)&Ks[col][half * 8];
        h8 hi = *(const h8*)&Ks[col][16 + half * 8];
#pragma unroll
        for (int e = 0; e < 8; ++e) { bk[e] = lo[e]; bk[8 + e] = hi[e]; }
        v8f z;
#pragma unroll
        for (int e = 0; e < 8; ++e) z[e] = 0.0f;
        sc[t] = __builtin_amdgcn_wmma_f32_16x16x32_f16(false, aq, false, bk,
                                                       (short)0, z, false, false);
    }

    // --- bias + shift-mask + softmax over the wave's 16-row strip ---
    const float scale = 0.17677669529663687f;    // 1/sqrt(32)
    int i2[4], j2[4], labm[4]; bool mval[4];
#pragma unroll
    for (int t = 0; t < 4; ++t) {
        int m = t * 16 + ln;
        mval[t] = (m < kNTok);
        int ii = m / kWs, jj = m - ii * kWs;
        i2[t] = ii; j2[t] = jj;
        labm[t] = swin_band(wh * kWs + ii) * 3 + swin_band(ww * kWs + jj);
    }
    float sv[4][8], mx[8], sum[8];
#pragma unroll
    for (int e = 0; e < 8; ++e) mx[e] = -1e30f;
#pragma unroll
    for (int t = 0; t < 4; ++t)
#pragma unroll
        for (int e = 0; e < 8; ++e) {
            int n = strip + e + half * 8;        // query token
            float s = -1e30f;
            if (n < kNTok && mval[t]) {
                int ii = n / kWs, jj = n - ii * kWs;
                int labn = swin_band(wh * kWs + ii) * 3 + swin_band(ww * kWs + jj);
                int ridx = (ii - i2[t] + 6) * 13 + (jj - j2[t] + 6);
                s = sc[t][e] * scale + rpbh[ridx] + (labn != labm[t] ? -100.0f : 0.0f);
            }
            sv[t][e] = s;
            mx[e] = fmaxf(mx[e], s);
        }
#pragma unroll
    for (int m = 1; m <= 8; m <<= 1)
#pragma unroll
        for (int e = 0; e < 8; ++e) mx[e] = fmaxf(mx[e], __shfl_xor(mx[e], m, 32));
#pragma unroll
    for (int e = 0; e < 8; ++e) sum[e] = 0.0f;
#pragma unroll
    for (int t = 0; t < 4; ++t)
#pragma unroll
        for (int e = 0; e < 8; ++e) {
            float p = (sv[t][e] > -1e29f) ? __expf(sv[t][e] - mx[e]) : 0.0f;
            sv[t][e] = p;
            sum[e] += p;
        }
#pragma unroll
    for (int m = 1; m <= 8; m <<= 1)
#pragma unroll
        for (int e = 0; e < 8; ++e) sum[e] += __shfl_xor(sum[e], m, 32);
    float inv[8];
#pragma unroll
    for (int e = 0; e < 8; ++e) inv[e] = sum[e] > 0.0f ? 1.0f / sum[e] : 0.0f;
#pragma unroll
    for (int t = 0; t < 4; ++t)
#pragma unroll
        for (int e = 0; e < 8; ++e)
            Am[strip + e + half * 8][t * 16 + ln] = (_Float16)(sv[t][e] * inv[e]);
    __syncthreads();

    // --- stage 2: O = Am(16x64) @ V(64x32), two K steps of 32, two 16-col tiles
    v8f oacc[2];
#pragma unroll
    for (int nt = 0; nt < 2; ++nt)
#pragma unroll
        for (int e = 0; e < 8; ++e) oacc[nt][e] = 0.0f;
#pragma unroll
    for (int ks = 0; ks < 2; ++ks) {
        int kb = ks * 32;
        v16h aa;
        {
            int row = strip + ln;
            h8 lo = *(const h8*)&Am[row][kb + half * 8];
            h8 hi = *(const h8*)&Am[row][kb + 16 + half * 8];
#pragma unroll
            for (int e = 0; e < 8; ++e) { aa[e] = lo[e]; aa[8 + e] = hi[e]; }
        }
#pragma unroll
        for (int nt = 0; nt < 2; ++nt) {
            int col = nt * 16 + ln;              // output channel d
            v16h bv;
            h8 lo = *(const h8*)&Vt[col][kb + half * 8];
            h8 hi = *(const h8*)&Vt[col][kb + 16 + half * 8];
#pragma unroll
            for (int e = 0; e < 8; ++e) { bv[e] = lo[e]; bv[8 + e] = hi[e]; }
            oacc[nt] = __builtin_amdgcn_wmma_f32_16x16x32_f16(
                false, aa, false, bv, (short)0, oacc[nt], false, false);
        }
    }
#pragma unroll
    for (int nt = 0; nt < 2; ++nt)
#pragma unroll
        for (int e = 0; e < 8; ++e) {
            int n = strip + e + half * 8;
            if (n < kNTok)
                out[(rowbase + n) * kDim + h * kHd + nt * 16 + ln] = (_Float16)oacc[nt][e];
        }
}

// ---------------------------------------------------------------------------
extern "C" void kernel_launch(void* const* d_in, const int* in_sizes, int n_in,
                              void* d_out, int out_size, void* d_ws, size_t ws_size,
                              hipStream_t stream) {
    const float* x     = (const float*)d_in[0];
    const float* n1g   = (const float*)d_in[3];
    const float* n1b   = (const float*)d_in[4];
    const float* qkvw  = (const float*)d_in[5];
    const float* qkvb  = (const float*)d_in[6];
    const float* rpb   = (const float*)d_in[7];
    const float* projw = (const float*)d_in[8];
    const float* projb = (const float*)d_in[9];
    const float* n2g   = (const float*)d_in[10];
    const float* n2b   = (const float*)d_in[11];
    const float* f1w   = (const float*)d_in[12];
    const float* f1b   = (const float*)d_in[13];
    const float* f2w   = (const float*)d_in[14];
    const float* f2b   = (const float*)d_in[15];
    float* out = (float*)d_out;

    char* ws = (char*)d_ws;
    auto alloc = [&](size_t bytes) -> char* {
        char* p = ws;
        ws += (bytes + 255) & ~(size_t)255;
        return p;
    };
    _Float16* wqkv = (_Float16*)alloc((size_t)kDim * kQkvN * 2);
    _Float16* wprj = (_Float16*)alloc((size_t)kDim * kDim * 2);
    _Float16* w1   = (_Float16*)alloc((size_t)kDim * kMlpH * 2);
    _Float16* w2   = (_Float16*)alloc((size_t)kMlpH * kDim * 2);
    // region A (77 MB): xw -> attn_out -> h16 (lifetimes disjoint)
    char* regA = alloc((size_t)kRows * kDim * 2);
    // region B (231 MB): qkv out -> xres f32 (154 MB, lifetimes disjoint)
    char* regB = alloc((size_t)kRows * kQkvN * 2);
    // hidden (308 MB)
    _Float16* hid = (_Float16*)alloc((size_t)kRows * kMlpH * 2);

    _Float16* xw    = (_Float16*)regA;
    _Float16* attnh = (_Float16*)regA;
    _Float16* h16   = (_Float16*)regA;
    _Float16* qkvh  = (_Float16*)regB;
    float*    xres  = (float*)regB;

    // 1. weight casts -> f16 (layouts stay natural (N x K))
    cast_f16_kernel<<<(kQkvN * kDim + 255) / 256, 256, 0, stream>>>(qkvw, wqkv, kQkvN * kDim);
    cast_f16_kernel<<<(kDim * kDim + 255) / 256, 256, 0, stream>>>(projw, wprj, kDim * kDim);
    cast_f16_kernel<<<(kMlpH * kDim + 255) / 256, 256, 0, stream>>>(f1w, w1, kMlpH * kDim);
    cast_f16_kernel<<<(kDim * kMlpH + 255) / 256, 256, 0, stream>>>(f2w, w2, kDim * kMlpH);

    // 2. LN1 + shift + window-partition, f32 -> f16 windowed rows
    ln_to_f16_kernel<0><<<kRows / 8, 256, 0, stream>>>(x, n1g, n1b, xw);

    // 3. QKV GEMM: (100352 x 384) @ (384 x 1152) + b -> f16
    gemm_wmma_kernel<kDim, 0><<<dim3(kRows / 128, kQkvN / 128), 256, 0, stream>>>(
        xw, wqkv, qkvb, nullptr, qkvh, nullptr, kQkvN);

    // 4. windowed attention (2048 windows x 12 heads)
    attn_kernel<<<(kRows / kNTok) * kHeads, 128, 0, stream>>>(qkvh, rpb, attnh);

    // 5. proj GEMM + window-reverse + shortcut -> xres (f32, natural layout)
    gemm_wmma_kernel<kDim, 2><<<dim3(kRows / 128, kDim / 128), 256, 0, stream>>>(
        attnh, wprj, projb, x, nullptr, xres, kDim);

    // 6. LN2 -> f16
    ln_to_f16_kernel<1><<<kRows / 8, 256, 0, stream>>>(xres, n2g, n2b, h16);

    // 7. FC1 GEMM + exact GELU -> f16 hidden
    gemm_wmma_kernel<kDim, 1><<<dim3(kRows / 128, kMlpH / 128), 256, 0, stream>>>(
        h16, w1, f1b, nullptr, hid, nullptr, kMlpH);

    // 8. FC2 GEMM + residual(xres) -> d_out (f32)
    gemm_wmma_kernel<kMlpH, 3><<<dim3(kRows / 128, kDim / 128), 256, 0, stream>>>(
        hid, w2, f2b, xres, nullptr, out, kDim);

    (void)in_sizes; (void)n_in; (void)out_size; (void)ws_size;
}
